// mBP_model_10548439679434
// MI455X (gfx1250) — matrix operand order repaired and checked by reading, not compiled, not run on hardware
//
#include <hip/hip_runtime.h>
#include <hip/hip_bf16.h>

// ---------------- problem constants (match reference) ----------------
#define E_EDGES   65536
#define NAT       2048
#define NRAD      24
#define NCOMP     5          // 1 two-body + 4 zeta components
#define NK        69         // total lxlylz combos across zeta=1..4
#define NFEAT     216        // NRAD * (1 + 2*4)
#define EMB       16
#define FDIM      3456       // NFEAT * EMB
#define H1DIM     256
#define H2DIM     128
#define RCUT      6.0f

typedef __attribute__((ext_vector_type(16))) _Float16 v16h;
typedef __attribute__((ext_vector_type(8)))  float    v8f;

// ---------------- static lxlylz tables (reference enumeration order) ----
// packed lx | ly<<4 | lz<<8 ; CFN = z!/((z-n)! lx! ly! lz!) ; lambda=(-1)^n
// zeta offsets: [0,4) z=1, [4,14) z=2, [14,34) z=3, [34,69) z=4
#define PK(x,y,z) ((x) | ((y)<<4) | ((z)<<8))
static __device__ const int CPK[NK] = {
  // z=1
  PK(0,0,0), PK(0,0,1), PK(0,1,0), PK(1,0,0),
  // z=2
  PK(0,0,0), PK(0,0,1), PK(0,1,0), PK(1,0,0),
  PK(0,0,2), PK(0,1,1), PK(0,2,0), PK(1,0,1), PK(1,1,0), PK(2,0,0),
  // z=3
  PK(0,0,0), PK(0,0,1), PK(0,1,0), PK(1,0,0),
  PK(0,0,2), PK(0,1,1), PK(0,2,0), PK(1,0,1), PK(1,1,0), PK(2,0,0),
  PK(0,0,3), PK(0,1,2), PK(0,2,1), PK(0,3,0), PK(1,0,2), PK(1,1,1),
  PK(1,2,0), PK(2,0,1), PK(2,1,0), PK(3,0,0),
  // z=4
  PK(0,0,0), PK(0,0,1), PK(0,1,0), PK(1,0,0),
  PK(0,0,2), PK(0,1,1), PK(0,2,0), PK(1,0,1), PK(1,1,0), PK(2,0,0),
  PK(0,0,3), PK(0,1,2), PK(0,2,1), PK(0,3,0), PK(1,0,2), PK(1,1,1),
  PK(1,2,0), PK(2,0,1), PK(2,1,0), PK(3,0,0),
  PK(0,0,4), PK(0,1,3), PK(0,2,2), PK(0,3,1), PK(0,4,0),
  PK(1,0,3), PK(1,1,2), PK(1,2,1), PK(1,3,0),
  PK(2,0,2), PK(2,1,1), PK(2,2,0), PK(3,0,1), PK(3,1,0), PK(4,0,0)
};
static __device__ const float CFN[NK] = {
  // z=1
  1,1,1,1,
  // z=2
  1,2,2,2, 1,2,1,2,2,1,
  // z=3
  1,3,3,3, 3,6,3,6,6,3, 1,3,3,1,3,6,3,3,3,1,
  // z=4
  1,4,4,4, 6,12,6,12,12,6, 4,12,12,4,12,24,12,12,12,4,
  1,4,6,4,1, 4,12,12,4, 6,12,6, 4,4,1
};
static __device__ const int KOFF[5] = {0, 4, 14, 34, 69};

__device__ __forceinline__ float silu_f(float x) {
  return x / (1.0f + __expf(-x));
}
__device__ __forceinline__ float ipw(float b, int e) {
  float r = 1.0f;
  for (int i = 0; i < e; ++i) r *= b;
  return r;
}
__device__ __forceinline__ int lower_bound_seg(const int* __restrict__ s, int n, int v) {
  int lo = 0, hi = n;
  while (lo < hi) { int m = (lo + hi) >> 1; if (s[m] < v) lo = m + 1; else hi = m; }
  return lo;
}

// gfx1250 async memory->LDS DMA (ASYNCcnt-tracked), 16 bytes per lane.
// LDS destination address is the low 32 bits of the generic pointer.
__device__ __forceinline__ void async_ld_b128(unsigned lds_off, const void* gaddr) {
  asm volatile("global_load_async_to_lds_b128 %0, %1, off"
               :: "v"(lds_off), "v"(gaddr) : "memory");
}
__device__ __forceinline__ void wait_async0() {
  asm volatile("s_wait_asynccnt 0x0" ::: "memory");
}

// ---------------- K0: zero output + f32->f16 weight conversion ----------
__global__ __launch_bounds__(256) void prep_kernel(
    const float* __restrict__ Wa1, const float* __restrict__ Wa2,
    _Float16* __restrict__ Wa1h, _Float16* __restrict__ Wa2h,
    float* __restrict__ out) {
  int i = blockIdx.x * 256 + threadIdx.x;
  if (i == 0) out[0] = 0.0f;
  if (i < FDIM * H1DIM) Wa1h[i] = (_Float16)Wa1[i];
  if (i < H1DIM * H2DIM) Wa2h[i] = (_Float16)Wa2[i];
}

// ---------------- K1: per-edge radial MLP (VALU f32, weights in LDS) ----
__global__ __launch_bounds__(256) void edge_radial_kernel(
    const float* __restrict__ rij,
    const float* __restrict__ Wr1, const float* __restrict__ br1,
    const float* __restrict__ Wr2, const float* __restrict__ br2,
    float* __restrict__ radial /* [E,120] */) {
  __shared__ float sW1[NRAD * 64];
  __shared__ float sB1[64];
  __shared__ float sW2[64 * (NCOMP * NRAD)];
  __shared__ float sB2[NCOMP * NRAD];
  for (int i = threadIdx.x; i < NRAD * 64; i += 256) sW1[i] = Wr1[i];
  for (int i = threadIdx.x; i < 64; i += 256) sB1[i] = br1[i];
  for (int i = threadIdx.x; i < 64 * NCOMP * NRAD; i += 256) sW2[i] = Wr2[i];
  for (int i = threadIdx.x; i < NCOMP * NRAD; i += 256) sB2[i] = br2[i];
  __syncthreads();

  int e = blockIdx.x * 256 + threadIdx.x;
  if (e >= E_EDGES) return;
  float x = rij[3 * e], y = rij[3 * e + 1], z = rij[3 * e + 2];
  float r = sqrtf(x * x + y * y + z * z);
  float rc = fminf(r, RCUT);
  float fc = 0.5f * (__cosf(3.14159265358979f * rc / RCUT) + 1.0f);

  float basis[NRAD];
  #pragma unroll
  for (int i = 0; i < NRAD; ++i) {
    float c = RCUT * (float)i / (float)(NRAD - 1);
    float d = r - c;
    basis[i] = __expf(-d * d * 8.0f) * fc;   // 1/(2*(RCUT/NRAD)^2) = 8
  }
  float h[64];
  #pragma unroll 4
  for (int o = 0; o < 64; ++o) {
    float acc = sB1[o];
    #pragma unroll
    for (int i = 0; i < NRAD; ++i) acc += basis[i] * sW1[i * 64 + o];
    h[o] = silu_f(acc);
  }
  float* rd = radial + (size_t)e * (NCOMP * NRAD);
  #pragma unroll 2
  for (int o = 0; o < NCOMP * NRAD; ++o) {
    float acc = sB2[o];
    #pragma unroll 8
    for (int i = 0; i < 64; ++i) acc += h[i] * sW2[i * (NCOMP * NRAD) + o];
    rd[o] = silu_f(acc);
  }
}

// ---------------- K2: per-atom segment reduction + features + X(f16) ----
// One block per atom; first_atom_idx is sorted so segment bounds come from a
// binary search (no global atomics -- naive port would issue ~110M of them).
__global__ __launch_bounds__(256) void atom_kernel(
    const float* __restrict__ rij, const float* __restrict__ radial,
    const int* __restrict__ seg, const int* __restrict__ species,
    const float* __restrict__ Ws1, const float* __restrict__ bs1,
    const float* __restrict__ Ws2, const float* __restrict__ bs2,
    _Float16* __restrict__ X /* [NAT, FDIM] */) {
  const int atom = blockIdx.x;
  const int tid = threadIdx.x;
  __shared__ float sRad[NCOMP * NRAD];
  __shared__ float sGij[NK];
  __shared__ float sGi[NRAD * NK];   // 1656 accumulated gi values
  __shared__ float sFeat[NFEAT];
  __shared__ float sT[32];
  __shared__ float sEmb[EMB];

  // per-thread feature slots: f = i*256 + tid over [0, 1656)
  int fr[7], fk[7], fri[7];
  #pragma unroll
  for (int i = 0; i < 7; ++i) {
    int f = i * 256 + tid;
    if (f < NRAD * NK) {
      int rr = f / NK, k = f % NK;
      int iz = (k < 4) ? 0 : (k < 14) ? 1 : (k < 34) ? 2 : 3;
      fr[i] = rr * NK + k;
      fk[i] = k;
      fri[i] = rr * NCOMP + iz + 1;
    } else { fr[i] = -1; fk[i] = 0; fri[i] = 0; }
  }
  int myPk = 0; float myFn = 0.0f;
  if (tid < NK) { myPk = CPK[tid]; myFn = CFN[tid]; }

  float acc[7] = {0, 0, 0, 0, 0, 0, 0};
  float tb = 0.0f;

  const int lo = lower_bound_seg(seg, E_EDGES, atom);
  const int hi = lower_bound_seg(seg, E_EDGES, atom + 1);
  for (int e = lo; e < hi; ++e) {
    if (tid < NCOMP * NRAD) sRad[tid] = radial[(size_t)e * (NCOMP * NRAD) + tid];
    if (tid < NK) {
      float x = rij[3 * e], y = rij[3 * e + 1], z = rij[3 * e + 2];
      float rinv = rsqrtf(x * x + y * y + z * z);
      float ux = x * rinv + 1e-12f, uy = y * rinv + 1e-12f, uz = z * rinv + 1e-12f;
      sGij[tid] = myFn * ipw(ux, myPk & 15) * ipw(uy, (myPk >> 4) & 15)
                       * ipw(uz, (myPk >> 8) & 15);
    }
    __syncthreads();
    if (tid < NRAD) tb += sRad[tid * NCOMP];      // two-body component 0
    #pragma unroll
    for (int i = 0; i < 7; ++i)
      if (fr[i] >= 0) acc[i] += sRad[fri[i]] * sGij[fk[i]];
    __syncthreads();
  }
  #pragma unroll
  for (int i = 0; i < 7; ++i) if (fr[i] >= 0) sGi[fr[i]] = acc[i];
  if (tid < NRAD) sFeat[tid] = tb;
  __syncthreads();

  // feats: per (radial channel, zeta, lambda-sign)
  if (tid < NRAD * 8) {
    int rr = tid >> 3, comp = tid & 7;
    int iz = comp >> 1, sign = comp & 1;
    float norm = __builtin_amdgcn_ldexpf(1.0f, -iz);  // 2^(1-z) with z=iz+1
    float s = 0.0f;
    for (int k = KOFF[iz]; k < KOFF[iz + 1]; ++k) {
      float v = sGi[rr * NK + k];
      float lam = 1.0f;
      if (sign) {
        int pk = CPK[k];
        int n = (pk & 15) + ((pk >> 4) & 15) + ((pk >> 8) & 15);
        lam = (n & 1) ? -1.0f : 1.0f;
      }
      s += v * v * lam;
    }
    sFeat[NRAD + (2 * iz + sign) * NRAD + rr] = s * norm;
  }
  // species embedding: emb = silu(Ws1[sp] + bs1) @ Ws2 + bs2
  int sp = species[atom];
  if (tid < 32) sT[tid] = silu_f(Ws1[sp * 32 + tid] + bs1[tid]);
  __syncthreads();
  if (tid < EMB) {
    float s = bs2[tid];
    #pragma unroll
    for (int j = 0; j < 32; ++j) s += sT[j] * Ws2[j * EMB + tid];
    sEmb[tid] = s;
  }
  __syncthreads();
  // X row in f16: x[f*EMB + e] = feat[f] * emb[e]
  _Float16* xrow = X + (size_t)atom * FDIM;
  for (int idx = tid; idx < FDIM; idx += 256)
    xrow[idx] = (_Float16)(sFeat[idx >> 4] * sEmb[idx & 15]);
}

// ---------------- K3/K4: WMMA GEMM  out = act(A[M,K] @ B[K,N] + bias) ----
// 128x64 block tile, 8 waves in 4x2; each wave owns a 32x32 macro-tile:
// 2 A-frags x 2 B-frags -> 4 v_wmma_f32_16x16x32_f16 per K-step (each
// fragment reused twice). K-step 32, f32 accumulation.
// Staging: double-buffered global_load_async_to_lds_b128 (one 16B DMA per
// lane per tile: A 4KB, B 4KB), overlapped with the WMMAs;
// s_wait_asynccnt + barrier close each stage.
union Frag { v16h v; unsigned int u[8]; };

#define LDA 48   // 32 K-halfs + pad; 96B row stride (16B multiple)
#define LDB 80   // 64 N-halfs + pad; 160B row stride (16B multiple)

__device__ __forceinline__ void load_afrag(Frag& f, const _Float16 (*sa)[LDA],
                                           int mt, int lane) {
  // A 16x32 f16 layout: lane<16 -> row M=lane, K pairs {0..7,16..23};
  // lanes 16-31 -> same rows, K pairs {8..15,24..31}
  int half = lane >> 4, mrow = (lane & 15) + mt * 16;
  #pragma unroll
  for (int j = 0; j < 8; ++j) {
    int kp = ((j < 4) ? (2 * j) : (16 + 2 * (j - 4))) + 8 * half;
    f.u[j] = *(const unsigned int*)&sa[mrow][kp];
  }
}
__device__ __forceinline__ void load_bfrag(Frag& f, const _Float16 (*sb)[LDB],
                                           int nt, int lane) {
  // B 32x16 f16 layout: lane<16 -> col N=lane, K pairs {0..7,16..23};
  // lanes 16-31 -> same cols, K pairs {8..15,24..31}
  int half = lane >> 4, ncol = (lane & 15) + nt * 16;
  #pragma unroll
  for (int j = 0; j < 8; ++j) {
    int kp = ((j < 4) ? (2 * j) : (16 + 2 * (j - 4))) + 8 * half;
    f.v[2 * j] = sb[kp][ncol];
    f.v[2 * j + 1] = sb[kp + 1][ncol];
  }
}

template <bool SILU, bool OUT_HALF>
__global__ __launch_bounds__(256) void gemm_wmma(
    const _Float16* __restrict__ A, const _Float16* __restrict__ B,
    const float* __restrict__ bias, void* __restrict__ out,
    int M, int N, int K) {
  __shared__ _Float16 sA[2][128][LDA];
  __shared__ _Float16 sB[2][32][LDB];
  const int tid = threadIdx.x;
  const int wave = tid >> 5, lane = tid & 31;
  const int m0 = blockIdx.x * 128, n0 = blockIdx.y * 64;
  const int wm = wave >> 1, wn = wave & 1;       // 4x2 wave grid
  const int mt0 = 2 * wm, mt1 = 2 * wm + 1;
  const int nt0 = 2 * wn, nt1 = 2 * wn + 1;

  // per-thread staging coordinates (one b128 per lane per tile)
  const int arow = tid >> 1, ac = (tid & 1) * 8;  // 128 rows x 2 x 8 halfs
  const int brow = tid >> 3, bc = (tid & 7) * 8;  //  32 rows x 8 x 8 halfs

  auto stage = [&](int buf, int k0) {
    unsigned la = (unsigned)(uintptr_t)&sA[buf][arow][ac];
    async_ld_b128(la, A + (size_t)(m0 + arow) * K + k0 + ac);
    unsigned lb = (unsigned)(uintptr_t)&sB[buf][brow][bc];
    async_ld_b128(lb, B + (size_t)(k0 + brow) * N + n0 + bc);
  };

  v8f c00 = {}, c01 = {}, c10 = {}, c11 = {};

  stage(0, 0);
  wait_async0();
  __syncthreads();

  int cur = 0;
  for (int k0 = 0; k0 < K; k0 += 32) {
    Frag a0, a1, b0, b1;
    load_afrag(a0, sA[cur], mt0, lane);
    load_afrag(a1, sA[cur], mt1, lane);
    load_bfrag(b0, sB[cur], nt0, lane);
    load_bfrag(b1, sB[cur], nt1, lane);

    // kick off next tile's DMA; it overlaps with the WMMAs below
    if (k0 + 32 < K) stage(cur ^ 1, k0 + 32);

    c00 = __builtin_amdgcn_wmma_f32_16x16x32_f16(false, a0.v, false, b0.v,
                                                 (short)0, c00, false, false);
    c01 = __builtin_amdgcn_wmma_f32_16x16x32_f16(false, a0.v, false, b1.v,
                                                 (short)0, c01, false, false);
    c10 = __builtin_amdgcn_wmma_f32_16x16x32_f16(false, a1.v, false, b0.v,
                                                 (short)0, c10, false, false);
    c11 = __builtin_amdgcn_wmma_f32_16x16x32_f16(false, a1.v, false, b1.v,
                                                 (short)0, c11, false, false);

    wait_async0();       // our async writes for the next buffer are done
    __syncthreads();     // (compiler waits dscnt before signaling: frag reads done)
    cur ^= 1;
  }

  // epilogue: C/D layout lane(0-15)=N, lanes 16-31 are M+8; VGPR j = M row
  const int ncol = lane & 15, mbase = (lane >> 4) * 8;
  auto store_tile = [&](const v8f& c, int mt, int nt) {
    #pragma unroll
    for (int j = 0; j < 8; ++j) {
      int m = m0 + mt * 16 + mbase + j;
      int n = n0 + nt * 16 + ncol;
      float v = c[j] + bias[n];
      if (SILU) v = silu_f(v);
      if (OUT_HALF) ((_Float16*)out)[(size_t)m * N + n] = (_Float16)v;
      else          ((float*)out)[(size_t)m * N + n] = v;
    }
  };
  store_tile(c00, mt0, nt0);
  store_tile(c01, mt0, nt1);
  store_tile(c10, mt1, nt0);
  store_tile(c11, mt1, nt1);
}

// ---------------- K5: per-atom energy + wave-reduced global sum ---------
__global__ __launch_bounds__(256) void energy_kernel(
    const float* __restrict__ H2, const float* __restrict__ Wa3,
    const float* __restrict__ ba3, float* __restrict__ out) {
  int a = blockIdx.x * 256 + threadIdx.x;
  if (a >= NAT) return;
  float s = ba3[0];
  const float* h = H2 + (size_t)a * H2DIM;
  #pragma unroll 8
  for (int j = 0; j < H2DIM; ++j) s += h[j] * Wa3[j];
  // wave32 tree reduction, then one atomic per wave
  #pragma unroll
  for (int o = 16; o > 0; o >>= 1) s += __shfl_down(s, o, 32);
  if ((threadIdx.x & 31) == 0) atomicAdd(out, s);
}

// ---------------- launch ------------------------------------------------
extern "C" void kernel_launch(void* const* d_in, const int* in_sizes, int n_in,
                              void* d_out, int out_size, void* d_ws, size_t ws_size,
                              hipStream_t stream) {
  const float* rij = (const float*)d_in[0];
  const float* Wr1 = (const float*)d_in[1];
  const float* br1 = (const float*)d_in[2];
  const float* Wr2 = (const float*)d_in[3];
  const float* br2 = (const float*)d_in[4];
  const float* Ws1 = (const float*)d_in[5];
  const float* bs1 = (const float*)d_in[6];
  const float* Ws2 = (const float*)d_in[7];
  const float* bs2 = (const float*)d_in[8];
  const float* Wa1 = (const float*)d_in[9];
  const float* ba1 = (const float*)d_in[10];
  const float* Wa2 = (const float*)d_in[11];
  const float* ba2 = (const float*)d_in[12];
  const float* Wa3 = (const float*)d_in[13];
  const float* ba3 = (const float*)d_in[14];
  const int* seg = (const int*)d_in[15];
  const int* species = (const int*)d_in[16];
  float* out = (float*)d_out;

  char* ws = (char*)d_ws;
  size_t off = 0;
  auto take = [&](size_t bytes) -> void* {
    void* p = ws + off;
    off = (off + bytes + 255) & ~(size_t)255;
    return p;
  };
  float*     radial = (float*)take((size_t)E_EDGES * NCOMP * NRAD * 4); // 31.5 MB
  _Float16*  X      = (_Float16*)take((size_t)NAT * FDIM * 2);          // 14.2 MB
  _Float16*  Wa1h   = (_Float16*)take((size_t)FDIM * H1DIM * 2);        // 1.8 MB
  _Float16*  Wa2h   = (_Float16*)take((size_t)H1DIM * H2DIM * 2);       // 64 KB
  _Float16*  H1h    = (_Float16*)take((size_t)NAT * H1DIM * 2);         // 1 MB
  float*     H2     = (float*)take((size_t)NAT * H2DIM * 4);            // 1 MB

  prep_kernel<<<(FDIM * H1DIM + 255) / 256, 256, 0, stream>>>(Wa1, Wa2, Wa1h, Wa2h, out);
  edge_radial_kernel<<<E_EDGES / 256, 256, 0, stream>>>(rij, Wr1, br1, Wr2, br2, radial);
  atom_kernel<<<NAT, 256, 0, stream>>>(rij, radial, seg, species,
                                       Ws1, bs1, Ws2, bs2, X);
  gemm_wmma<true, true><<<dim3(NAT / 128, H1DIM / 64), 256, 0, stream>>>(
      X, Wa1h, ba1, (void*)H1h, NAT, H1DIM, FDIM);
  gemm_wmma<true, false><<<dim3(NAT / 128, H2DIM / 64), 256, 0, stream>>>(
      H1h, Wa2h, ba2, (void*)H2, NAT, H2DIM, H1DIM);
  energy_kernel<<<NAT / 256, 256, 0, stream>>>(H2, Wa3, ba3, out);
}
